// TLS_MSC_32444182954666
// MI455X (gfx1250) — matrix operand order repaired and checked by reading, not compiled, not run on hardware
//
#include <hip/hip_runtime.h>
#include <hip/hip_bf16.h>
#include <stdint.h>

#define HWPIX 65536
#define BIN   200
#define KPAD  224
#define DDIM  128
#define NSP   1024
#define NEG_SLOPE 0.01f

typedef __attribute__((ext_vector_type(16))) __bf16 v16bf;
typedef __attribute__((ext_vector_type(8)))  __bf16 v8bf;
typedef __attribute__((ext_vector_type(8)))  float  v8f;

static __device__ __forceinline__ v16bf join16(v8bf lo, v8bf hi) {
  return __builtin_shufflevector(lo, hi, 0,1,2,3,4,5,6,7,8,9,10,11,12,13,14,15);
}
static __device__ __forceinline__ v8bf ld8(const __bf16* p) { return *(const v8bf*)p; }
static __device__ __forceinline__ float lrelu(float v){ return v > 0.f ? v : NEG_SLOPE * v; }
static __device__ __forceinline__ v8f wmma_bf16(v16bf a, v16bf b, v8f c) {
  return __builtin_amdgcn_wmma_f32_16x16x32_bf16(false, a, false, b, (short)0, c, false, false);
}
union PK8 { v8bf v; uint4 u; };

// ---- CDNA5 async global->LDS (16B per lane, GVS mode), tracked by ASYNCcnt ----
static __device__ __forceinline__ void async_ld16(unsigned lds_off, const void* base, int voff){
  asm volatile("global_load_async_to_lds_b128 %0, %1, %2"
               :: "v"(lds_off), "v"(voff), "s"(base) : "memory");
}
static __device__ __forceinline__ void wait_async0(){
  asm volatile("s_wait_asynccnt 0" ::: "memory");
}
// low 32 bits of a flat LDS pointer are the LDS byte offset (ISA aperture rule)
static __device__ __forceinline__ unsigned lds_off32(const void* p){
  return (unsigned)(unsigned long long)p;
}

// ---------------- small prep kernels ----------------

__global__ void k_zero(float* p, int n){
  int i = blockIdx.x*256 + threadIdx.x;
  if (i < n) p[i] = 0.f;
}

// Fold BN into weights; build transposed bf16 weight matrices.
__global__ void k_weights(const float* bn0_g, const float* bn0_b, const float* w1, const float* b1,
                          const float* bn1_g, const float* bn1_b, const float* w2, const float* b2,
                          const float* gcn_w,
                          __bf16* w1pT, float* b1p, __bf16* w2pT, float* b2p, __bf16* gwT){
  int tid = threadIdx.x;
  for (int idx = tid; idx < DDIM*KPAD; idx += 256){
    int e = idx / KPAD, k = idx % KPAD;
    float v = (k < BIN) ? bn0_g[k] * w1[k*DDIM + e] : 0.f;
    w1pT[e*KPAD + k] = (__bf16)v;
  }
  for (int e = tid; e < DDIM; e += 256){
    float s = b1[e];
    for (int k = 0; k < BIN; ++k) s += bn0_b[k] * w1[k*DDIM + e];
    b1p[e] = s;
  }
  for (int idx = tid; idx < DDIM*DDIM; idx += 256){
    int e = idx / DDIM, k = idx % DDIM;
    w2pT[e*DDIM + k] = (__bf16)(bn1_g[k] * w2[k*DDIM + e]);
  }
  for (int e = tid; e < DDIM; e += 256){
    float s = b2[e];
    for (int k = 0; k < DDIM; ++k) s += bn1_b[k] * w2[k*DDIM + e];
    b2p[e] = s;
  }
  for (int idx = tid; idx < 3*DDIM*DDIM; idx += 256){
    int l = idx / (DDIM*DDIM), r = idx % (DDIM*DDIM);
    int e = r / DDIM, k = r % DDIM;
    gwT[l*DDIM*DDIM + e*DDIM + k] = (__bf16)gcn_w[l*DDIM*DDIM + k*DDIM + e];
  }
}

// column sums of Q (for normQ): each block reduces 128 rows locally, then atomics.
__global__ void k_colsum(const float* Q, float* qsum){
  int r0 = blockIdx.x * 128;
  int t = threadIdx.x;
  float a0=0.f, a1=0.f, a2=0.f, a3=0.f;
  for (int rr = 0; rr < 128; ++rr){
    const float* row = Q + (size_t)(r0+rr)*NSP;
    a0 += row[t]; a1 += row[t+256]; a2 += row[t+512]; a3 += row[t+768];
  }
  atomicAdd(qsum + t,       a0);
  atomicAdd(qsum + t + 256, a1);
  atomicAdd(qsum + t + 512, a2);
  atomicAdd(qsum + t + 768, a3);
}

// dinv[r] = rsqrt(rowsum(adj)+1)  (g = adj + I)
__global__ void k_dinv(const float* adj, float* dinv){
  __shared__ float red[256];
  int r = blockIdx.x, t = threadIdx.x;
  const float* row = adj + (size_t)r*NSP;
  float s = 0.f;
  for (int c = t; c < NSP; c += 256) s += row[c];
  red[t] = s; __syncthreads();
  for (int w = 128; w > 0; w >>= 1){ if (t < w) red[t] += red[t+w]; __syncthreads(); }
  if (t == 0) dinv[r] = rsqrtf(red[0] + 1.0f);
}

// gnb = bf16( dinv_r * (adj + I) * dinv_c )
__global__ void k_gnb(const float* adj, const float* dinv, __bf16* gnb){
  int idx = blockIdx.x*256 + threadIdx.x;
  int r = idx >> 10, c = idx & 1023;
  float v = adj[idx] + (r == c ? 1.f : 0.f);
  gnb[idx] = (__bf16)(dinv[r] * v * dinv[c]);
}

// hbT0[d][n] = bn_g0[d]*(h_raw[n][d]/qsum[n]) + bn_b0[d]
__global__ void k_hb0(const float* h_raw, const float* qsum, const float* g0, const float* b0,
                      __bf16* hbT){
  int idx = blockIdx.x*256 + threadIdx.x;
  int d = idx >> 10, n = idx & 1023;
  float h = h_raw[(size_t)n*DDIM + d] / qsum[n];
  hbT[idx] = (__bf16)(g0[d]*h + b0[d]);
}

// ---------------- WMMA GEMM kernels ----------------

// CNN: per 128-pixel tile: T1 = lrelu(X@W1'+b1'); XF = lrelu(T1@W2'+b2') -> xfT bf16 [D][HW]
__global__ void __launch_bounds__(256) k_cnn(const float* x, const __bf16* w1pT, const float* b1p,
                                             const __bf16* w2pT, const float* b2p, __bf16* xfT){
  extern __shared__ __align__(16) char smem[];
  __bf16* s_x = (__bf16*)smem;                 // 128 x 224
  __bf16* s_w = (__bf16*)(smem + 57344);       // 128 x 224 (W1'T), later 128 x 128 (W2'T)
  __bf16* s_t = (__bf16*)(smem + 114688);      // 128 x 128
  const int tid = threadIdx.x;
  const int lane = tid & 31, wv = tid >> 5;
  const int r16 = lane & 15, hf = lane >> 4;
  const int pix0 = blockIdx.x * 128;
  const unsigned swOff = lds_off32(s_w);

  // stage W1'T asynchronously (57344 bytes = 3584 x 16B), overlap with X conversion
  for (int i = tid; i < 3584; i += 256)
    async_ld16(swOff + (unsigned)i*16u, w1pT, i*16);

  for (int idx = tid; idx < 128*KPAD; idx += 256){
    int rr = idx / KPAD, c = idx % KPAD;
    float v = (c < BIN) ? x[(size_t)(pix0+rr)*BIN + c] : 0.f;
    s_x[idx] = (__bf16)v;
  }
  wait_async0();
  __syncthreads();

  const v8f vz = {0.f,0.f,0.f,0.f,0.f,0.f,0.f,0.f};
  v8f acc[8];
  for (int nt = 0; nt < 8; ++nt) acc[nt] = vz;
  const int arow = 16*wv + r16;

  for (int ks = 0; ks < 7; ++ks){
    const __bf16* ap = s_x + arow*KPAD + ks*32 + hf*8;
    v16bf a = join16(ld8(ap), ld8(ap + 16));
    for (int nt = 0; nt < 8; ++nt){
      const __bf16* bp = s_w + (nt*16 + r16)*KPAD + ks*32 + hf*8;
      acc[nt] = wmma_bf16(a, join16(ld8(bp), ld8(bp + 16)), acc[nt]);
    }
  }
  __syncthreads();
  for (int nt = 0; nt < 8; ++nt){
    int n = nt*16 + r16;
    float bias = b1p[n];
    for (int i = 0; i < 8; ++i){
      int row = 16*wv + 8*hf + i;
      s_t[row*DDIM + n] = (__bf16)lrelu(acc[nt][i] + bias);
    }
  }
  // stage W2'T asynchronously (32768 bytes = 2048 x 16B)
  for (int i = tid; i < 2048; i += 256)
    async_ld16(swOff + (unsigned)i*16u, w2pT, i*16);
  wait_async0();
  __syncthreads();

  for (int nt = 0; nt < 8; ++nt) acc[nt] = vz;
  for (int ks = 0; ks < 4; ++ks){
    const __bf16* ap = s_t + arow*DDIM + ks*32 + hf*8;
    v16bf a = join16(ld8(ap), ld8(ap + 16));
    for (int nt = 0; nt < 8; ++nt){
      const __bf16* bp = s_w + (nt*16 + r16)*DDIM + ks*32 + hf*8;
      acc[nt] = wmma_bf16(a, join16(ld8(bp), ld8(bp + 16)), acc[nt]);
    }
  }
  for (int nt = 0; nt < 8; ++nt){
    int d = nt*16 + r16;
    float bias = b2p[d];
    PK8 pk;
    for (int i = 0; i < 8; ++i) pk.v[i] = (__bf16)lrelu(acc[nt][i] + bias);
    int p = pix0 + 16*wv + 8*hf;
    *(uint4*)(xfT + (size_t)d*HWPIX + p) = pk.u;   // 8 consecutive pixels, 16B store
  }
}

// pooling: h_raw += (Q^T @ xf) partial over K-chunk (split-K with fp32 atomics)
__global__ void __launch_bounds__(256) k_pool(const float* Q, const __bf16* xfT, float* h_raw){
  extern __shared__ __align__(16) char smem[];
  __bf16* s_a = (__bf16*)smem;            // 128 x 40 (padded, transposed Q tile)
  __bf16* s_b = (__bf16*)(smem + 10240);  // 128 x 32
  const int tid = threadIdx.x, lane = tid & 31, wv = tid >> 5;
  const int r16 = lane & 15, hf = lane >> 4;
  const int m0 = blockIdx.x * 128;
  const int kbase = blockIdx.y * 2048;
  const unsigned sbOff = lds_off32(s_b);

  const v8f vz = {0.f,0.f,0.f,0.f,0.f,0.f,0.f,0.f};
  v8f acc[8];
  for (int nt = 0; nt < 8; ++nt) acc[nt] = vz;

  for (int kk = 0; kk < 64; ++kk){
    int kc = kbase + kk*32;
    // async-stage xf tile (bf16 -> LDS direct, 512 x 16B)
    for (int j = 0; j < 2; ++j){
      int c = tid + 256*j;
      int d = c >> 2, k8 = (c & 3) * 8;
      async_ld16(sbOff + (unsigned)(d*32 + k8)*2u, xfT,
                 (int)(((size_t)d*HWPIX + kc + k8) * 2));
    }
    // VALU-stage transposed Q tile with fp32->bf16 conversion (4096 floats)
    for (int j = 0; j < 4; ++j){
      int c = tid + 256*j;
      int k = c >> 5;
      int m4 = (c & 31) * 4;
      float4 v = *(const float4*)(Q + (size_t)(kc + k)*NSP + m0 + m4);
      s_a[(m4+0)*40 + k] = (__bf16)v.x;
      s_a[(m4+1)*40 + k] = (__bf16)v.y;
      s_a[(m4+2)*40 + k] = (__bf16)v.z;
      s_a[(m4+3)*40 + k] = (__bf16)v.w;
    }
    wait_async0();
    __syncthreads();
    const __bf16* ap = s_a + (16*wv + r16)*40 + hf*8;
    v16bf a = join16(ld8(ap), ld8(ap + 16));
    for (int nt = 0; nt < 8; ++nt){
      const __bf16* bp = s_b + (nt*16 + r16)*32 + hf*8;
      acc[nt] = wmma_bf16(a, join16(ld8(bp), ld8(bp + 16)), acc[nt]);
    }
    __syncthreads();
  }
  for (int nt = 0; nt < 8; ++nt){
    int d = nt*16 + r16;
    for (int i = 0; i < 8; ++i){
      int m = m0 + 16*wv + 8*hf + i;
      atomicAdd(h_raw + (size_t)m*DDIM + d, acc[nt][i]);
    }
  }
}

// one GCN layer: h' = lrelu((gn @ hb) @ W + b); writes next layer's hbT (BN fused) or final h3T.
__global__ void __launch_bounds__(256) k_gcn(const __bf16* gnb, const __bf16* hbT_in,
                                             const __bf16* gwT, const float* gbias,
                                             const float* ng, const float* nb, int has_next,
                                             __bf16* outT){
  extern __shared__ __align__(16) char smem[];
  __bf16* s_a = (__bf16*)smem;                    // 128 x 32
  __bf16* s_b = (__bf16*)(smem + 8192);           // 128 x 32
  __bf16* s_m = (__bf16*)(smem + 16384);          // 128 x 128 (m1 = gn@hb, bf16)
  __bf16* s_w = (__bf16*)(smem + 49152);          // 128 x 128 (W^T)
  const int tid = threadIdx.x, lane = tid & 31, wv = tid >> 5;
  const int r16 = lane & 15, hf = lane >> 4;
  const int m0 = blockIdx.x * 128;
  const int arow = 16*wv + r16;
  const unsigned saOff = lds_off32(s_a);
  const unsigned sbOff = lds_off32(s_b);
  const unsigned swOff = lds_off32(s_w);

  // prefetch W^T for phase 2 early (2048 x 16B) -- completes before phase-2 wait
  for (int i = tid; i < 2048; i += 256)
    async_ld16(swOff + (unsigned)i*16u, gwT, i*16);

  const v8f vz = {0.f,0.f,0.f,0.f,0.f,0.f,0.f,0.f};
  v8f acc[8];
  for (int nt = 0; nt < 8; ++nt) acc[nt] = vz;

  for (int kk = 0; kk < 32; ++kk){
    for (int j = 0; j < 2; ++j){
      int c = tid + 256*j;
      int r = c >> 2, off = (c & 3) * 8;
      async_ld16(saOff + (unsigned)(r*32 + off)*2u, gnb,
                 (int)(((size_t)(m0 + r)*NSP + kk*32 + off) * 2));
    }
    for (int j = 0; j < 2; ++j){
      int c = tid + 256*j;
      int d = c >> 2, off = (c & 3) * 8;
      async_ld16(sbOff + (unsigned)(d*32 + off)*2u, hbT_in,
                 (int)(((size_t)d*NSP + kk*32 + off) * 2));
    }
    wait_async0();
    __syncthreads();
    const __bf16* ap = s_a + arow*32 + hf*8;
    v16bf a = join16(ld8(ap), ld8(ap + 16));
    for (int nt = 0; nt < 8; ++nt){
      const __bf16* bp = s_b + (nt*16 + r16)*32 + hf*8;
      acc[nt] = wmma_bf16(a, join16(ld8(bp), ld8(bp + 16)), acc[nt]);
    }
    __syncthreads();
  }
  for (int nt = 0; nt < 8; ++nt){
    int d = nt*16 + r16;
    for (int i = 0; i < 8; ++i)
      s_m[(16*wv + 8*hf + i)*DDIM + d] = (__bf16)acc[nt][i];
  }
  __syncthreads();

  for (int nt = 0; nt < 8; ++nt) acc[nt] = vz;
  for (int ks = 0; ks < 4; ++ks){
    const __bf16* ap = s_m + arow*DDIM + ks*32 + hf*8;
    v16bf a = join16(ld8(ap), ld8(ap + 16));
    for (int nt = 0; nt < 8; ++nt){
      const __bf16* bp = s_w + (nt*16 + r16)*DDIM + ks*32 + hf*8;
      acc[nt] = wmma_bf16(a, join16(ld8(bp), ld8(bp + 16)), acc[nt]);
    }
  }
  for (int nt = 0; nt < 8; ++nt){
    int e = nt*16 + r16;
    float bias = gbias[e];
    float gg = has_next ? ng[e] : 1.f;
    float bb = has_next ? nb[e] : 0.f;
    PK8 pk;
    for (int i = 0; i < 8; ++i){
      float v = lrelu(acc[nt][i] + bias);
      pk.v[i] = (__bf16)(gg*v + bb);
    }
    int nbase = m0 + 16*wv + 8*hf;
    *(uint4*)(outT + (size_t)e*NSP + nbase) = pk.u;
  }
}

// out = Q @ h3  (A = Q row-major, B = h3T)
__global__ void __launch_bounds__(256) k_out(const float* Q, const __bf16* h3T, float* out){
  extern __shared__ __align__(16) char smem[];
  __bf16* s_a = (__bf16*)smem;           // 128 x 32
  __bf16* s_b = (__bf16*)(smem + 8192);  // 128 x 32
  const int tid = threadIdx.x, lane = tid & 31, wv = tid >> 5;
  const int r16 = lane & 15, hf = lane >> 4;
  const int p0 = blockIdx.x * 128;
  const unsigned sbOff = lds_off32(s_b);

  const v8f vz = {0.f,0.f,0.f,0.f,0.f,0.f,0.f,0.f};
  v8f acc[8];
  for (int nt = 0; nt < 8; ++nt) acc[nt] = vz;

  for (int kk = 0; kk < 32; ++kk){
    for (int j = 0; j < 2; ++j){
      int c = tid + 256*j;
      int d = c >> 2, off = (c & 3) * 8;
      async_ld16(sbOff + (unsigned)(d*32 + off)*2u, h3T,
                 (int)(((size_t)d*NSP + kk*32 + off) * 2));
    }
    for (int j = 0; j < 4; ++j){
      int c = tid + 256*j;
      int r = c >> 3, off = (c & 7) * 4;
      float4 v = *(const float4*)(Q + (size_t)(p0 + r)*NSP + kk*32 + off);
      s_a[r*32 + off + 0] = (__bf16)v.x;
      s_a[r*32 + off + 1] = (__bf16)v.y;
      s_a[r*32 + off + 2] = (__bf16)v.z;
      s_a[r*32 + off + 3] = (__bf16)v.w;
    }
    wait_async0();
    __syncthreads();
    const __bf16* ap = s_a + (16*wv + r16)*32 + hf*8;
    v16bf a = join16(ld8(ap), ld8(ap + 16));
    for (int nt = 0; nt < 8; ++nt){
      const __bf16* bp = s_b + (nt*16 + r16)*32 + hf*8;
      acc[nt] = wmma_bf16(a, join16(ld8(bp), ld8(bp + 16)), acc[nt]);
    }
    __syncthreads();
  }
  for (int nt = 0; nt < 8; ++nt){
    int d = nt*16 + r16;
    for (int i = 0; i < 8; ++i){
      int p = p0 + 16*wv + 8*hf + i;
      out[(size_t)p*DDIM + d] = acc[nt][i];
    }
  }
}

// ---------------- host-side launcher ----------------

extern "C" void kernel_launch(void* const* d_in, const int* in_sizes, int n_in,
                              void* d_out, int out_size, void* d_ws, size_t ws_size,
                              hipStream_t stream){
  (void)in_sizes; (void)n_in; (void)out_size; (void)ws_size;
  const float* x      = (const float*)d_in[0];
  const float* Q      = (const float*)d_in[1];
  const float* adj    = (const float*)d_in[2];
  const float* bn0_g  = (const float*)d_in[3];
  const float* bn0_b  = (const float*)d_in[4];
  const float* w1     = (const float*)d_in[5];
  const float* b1     = (const float*)d_in[6];
  const float* bn1_g  = (const float*)d_in[7];
  const float* bn1_b  = (const float*)d_in[8];
  const float* w2     = (const float*)d_in[9];
  const float* b2     = (const float*)d_in[10];
  const float* gbn_g  = (const float*)d_in[11];
  const float* gbn_b  = (const float*)d_in[12];
  const float* gcn_w  = (const float*)d_in[13];
  const float* gcn_b  = (const float*)d_in[14];
  float* out = (float*)d_out;

  char* ws = (char*)d_ws;
  __bf16* xfT   = (__bf16*)(ws + 0);           // 128*65536 bf16   = 16 MiB
  __bf16* gnb   = (__bf16*)(ws + 16777216);    // 1024*1024 bf16   = 2 MiB
  __bf16* hbA   = (__bf16*)(ws + 18874368);    // 128*1024 bf16
  __bf16* hbB   = (__bf16*)(ws + 19136512);
  __bf16* h3T   = (__bf16*)(ws + 19398656);
  float*  h_raw = (float*)(ws + 19660800);     // 1024*128 f32
  float*  qsum  = (float*)(ws + 20185088);     // 1024 f32 (contiguous after h_raw)
  float*  dinv  = (float*)(ws + 20189184);     // 1024 f32
  __bf16* w1pT  = (__bf16*)(ws + 20193280);    // 128*224 bf16
  __bf16* w2pT  = (__bf16*)(ws + 20250624);    // 128*128 bf16
  __bf16* gwT   = (__bf16*)(ws + 20283392);    // 3*128*128 bf16
  float*  b1p   = (float*)(ws + 20381696);     // 128 f32
  float*  b2p   = (float*)(ws + 20382208);     // 128 f32

  // h_raw and qsum are contiguous: zero both in one pass (132096 floats)
  k_zero<<<516, 256, 0, stream>>>(h_raw, 131072 + 1024);
  k_weights<<<1, 256, 0, stream>>>(bn0_g, bn0_b, w1, b1, bn1_g, bn1_b, w2, b2, gcn_w,
                                   w1pT, b1p, w2pT, b2p, gwT);
  k_colsum<<<512, 256, 0, stream>>>(Q, qsum);
  k_dinv<<<1024, 256, 0, stream>>>(adj, dinv);
  k_gnb<<<4096, 256, 0, stream>>>(adj, dinv, gnb);

  k_cnn<<<512, 256, 147456, stream>>>(x, w1pT, b1p, w2pT, b2p, xfT);
  k_pool<<<dim3(8, 32), 256, 18432, stream>>>(Q, xfT, h_raw);
  k_hb0<<<512, 256, 0, stream>>>(h_raw, qsum, gbn_g + 0, gbn_b + 0, hbA);

  k_gcn<<<8, 256, 81920, stream>>>(gnb, hbA, gwT,                 gcn_b,       gbn_g + 128, gbn_b + 128, 1, hbB);
  k_gcn<<<8, 256, 81920, stream>>>(gnb, hbB, gwT + DDIM*DDIM,     gcn_b + 128, gbn_g + 256, gbn_b + 256, 1, hbA);
  k_gcn<<<8, 256, 81920, stream>>>(gnb, hbA, gwT + 2*DDIM*DDIM,   gcn_b + 256, nullptr,     nullptr,     0, h3T);

  k_out<<<512, 256, 16384, stream>>>(Q, h3T, out);
}
